// GFA_72464688218296
// MI455X (gfx1250) — matrix operand order repaired and checked by reading, not compiled, not run on hardware
//
#include <hip/hip_runtime.h>
#include <stdint.h>

// ---------------- types for WMMA (CDNA5 gfx1250, wave32) ----------------
typedef __attribute__((ext_vector_type(16))) __bf16     v16bf;
typedef __attribute__((ext_vector_type(8)))  float      v8f;
typedef __attribute__((ext_vector_type(4)))  unsigned   u32x4;

union FragU  { u32x4 q[2]; v16bf v; };
union Half8  { unsigned short u[8]; u32x4 q; };

__device__ __forceinline__ unsigned short f2bf(float f) {
    union { float f; uint32_t u; } x; x.f = f;
    uint32_t r = x.u + 0x7FFFu + ((x.u >> 16) & 1u);   // round-to-nearest-even
    return (unsigned short)(r >> 16);
}

__device__ __forceinline__ v8f wmma_bf16(v16bf a, v16bf b, v8f c) {
    return __builtin_amdgcn_wmma_f32_16x16x32_bf16(false, a, false, b, (short)0, c, false, false);
}

// A fragment (16x32 bf16, M x K). src row-major [row][ld], row = mbase+(lane&15).
// Layout used here (contiguous k, same as B): lane L holds row M=L%16; K = khalf*16 + e, khalf=L>>4.
__device__ __forceinline__ v16bf load_a_frag(const unsigned short* base, int mrow,
                                             int ld, int kbase, int lane) {
    int khalf = lane >> 4;
    const unsigned short* r0 = base + mrow * ld + kbase + khalf * 16;
    FragU f;
    f.q[0] = *(const u32x4*)(r0);        // K = khalf*16 + 0..7
    f.q[1] = *(const u32x4*)(r0 + 8);    // K = khalf*16 + 8..15
    return f.v;
}

// B fragment (32x16 bf16, K x N). src row-major [nrow][ld] holding B^T (i.e. [n][k]).
// Layout used here (contiguous k, same as A): lane L holds column N=L%16; lanes 0-15: K=0..15, lanes 16-31: K=16..31.
__device__ __forceinline__ v16bf load_b_frag(const unsigned short* base, int nrow,
                                             int ld, int kbase, int lane) {
    int khalf = lane >> 4;
    const unsigned short* r0 = base + nrow * ld + kbase + khalf * 16;
    FragU f;
    f.q[0] = *(const u32x4*)(r0);        // K = khalf*16 + 0..7
    f.q[1] = *(const u32x4*)(r0 + 8);    // K = khalf*16 + 8..15
    return f.v;
}

// Y = relu(bn_scale * (W @ X) + bn_shift), W bf16 [256][256], X in LDS as xs_t[128][256]
// (pos-major). 8 waves cover 16 o-tiles x 8 pos-tiles.
// Register blocking: 2 o-tiles x 4 pos-tiles (64 acc VGPRs), two half passes.
//   trans=true  -> dst[pos][ch] (ld 256), vectorized 16B stores
//   trans=false -> dst[ch][pos] (ld 128)
__device__ __forceinline__ void project(const unsigned short* __restrict__ W,
                                        const float* __restrict__ bn,
                                        const unsigned short* xs_t,
                                        unsigned short* dst, bool trans,
                                        int wave, int lane)
{
    const int lm = lane & 15;
#pragma unroll 1
    for (int ph = 0; ph < 2; ++ph) {                 // pos-tile halves: 4 tiles each
        v8f acc[2][4];
#pragma unroll
        for (int oi = 0; oi < 2; ++oi)
#pragma unroll
            for (int pi = 0; pi < 4; ++pi) acc[oi][pi] = (v8f){};

#pragma unroll 2
        for (int kk = 0; kk < 8; ++kk) {
            v16bf a0 = load_a_frag(W, (wave * 2 + 0) * 16 + lm, 256, kk * 32, lane);
            v16bf a1 = load_a_frag(W, (wave * 2 + 1) * 16 + lm, 256, kk * 32, lane);
#pragma unroll
            for (int pi = 0; pi < 4; ++pi) {
                v16bf bv = load_b_frag(xs_t, (ph * 4 + pi) * 16 + lm, 256, kk * 32, lane);
                acc[0][pi] = wmma_bf16(a0, bv, acc[0][pi]);
                acc[1][pi] = wmma_bf16(a1, bv, acc[1][pi]);
            }
        }

        // epilogue: all loops fully unrolled so acc[][] is only indexed by constants
#pragma unroll
        for (int oi = 0; oi < 2; ++oi) {
            int ot = wave * 2 + oi;
            int obase = ot * 16 + (lane >> 4) * 8;
            float sc[8], sh[8];
#pragma unroll
            for (int r = 0; r < 8; ++r) {
                int o = obase + r;
                float g = bn[o], bb = bn[256 + o], mm = bn[512 + o], vv = bn[768 + o];
                float is = rsqrtf(vv + 1e-5f);
                sc[r] = g * is;
                sh[r] = bb - g * mm * is;
            }
#pragma unroll
            for (int pi = 0; pi < 4; ++pi) {
                int n = (ph * 4 + pi) * 16 + lm;
                Half8 h;
#pragma unroll
                for (int r = 0; r < 8; ++r) {
                    float y = acc[oi][pi][r] * sc[r] + sh[r];
                    h.u[r] = f2bf(y > 0.f ? y : 0.f);
                }
                if (trans) {
                    *(u32x4*)(dst + n * 256 + obase) = h.q;
                } else {
#pragma unroll
                    for (int r = 0; r < 8; ++r) dst[(obase + r) * 128 + n] = h.u[r];
                }
            }
        }
    }
}

// ---------------- GCB (tiny) ----------------
__global__ void gcb_logits(const float* __restrict__ feat, const float* __restrict__ W1,
                           const float* __restrict__ bn1, float* __restrict__ logits)
{
    int b = blockIdx.y;
    int n = blockIdx.x * 256 + threadIdx.x;              // grid.x = 64
    const float* fb = feat + (size_t)b * 256 * 16384;
    float acc = 0.f;
    for (int c = 0; c < 256; ++c) acc += W1[c] * fb[(size_t)c * 16384 + n];
    float g = bn1[0], bb = bn1[1], mm = bn1[2], vv = bn1[3];
    float is = rsqrtf(vv + 1e-5f);
    float y = acc * (g * is) + (bb - g * mm * is);
    logits[b * 16384 + n] = y > 0.f ? y : 0.f;
}

__global__ void gcb_softmax(float* __restrict__ logits)
{
    __shared__ float red[256];
    int b = blockIdx.x, tid = threadIdx.x;
    float* l = logits + b * 16384;
    float m = -3.4e38f;
    for (int n = tid; n < 16384; n += 256) m = fmaxf(m, l[n]);
    red[tid] = m; __syncthreads();
    for (int s = 128; s > 0; s >>= 1) { if (tid < s) red[tid] = fmaxf(red[tid], red[tid + s]); __syncthreads(); }
    m = red[0]; __syncthreads();
    float s = 0.f;
    for (int n = tid; n < 16384; n += 256) s += __expf(l[n] - m);
    red[tid] = s; __syncthreads();
    for (int st = 128; st > 0; st >>= 1) { if (tid < st) red[tid] += red[tid + st]; __syncthreads(); }
    float inv = 1.f / red[0];
    for (int n = tid; n < 16384; n += 256) l[n] = __expf(l[n] - m) * inv;
}

__global__ void gcb_gf(const float* __restrict__ feat, const float* __restrict__ att,
                       float* __restrict__ gf)
{
    __shared__ float red[256];
    int b = blockIdx.y, c = blockIdx.x, tid = threadIdx.x;
    const float* fr = feat + ((size_t)(b * 256 + c)) * 16384;
    const float* ar = att + b * 16384;
    float s = 0.f;
    for (int n = tid; n < 16384; n += 256) s += fr[n] * ar[n];
    red[tid] = s; __syncthreads();
    for (int st = 128; st > 0; st >>= 1) { if (tid < st) red[tid] += red[tid + st]; __syncthreads(); }
    if (tid == 0) gf[b * 256 + c] = red[0];
}

__global__ void gcb_gate(const float* __restrict__ gf, const float* __restrict__ fc1,
                         const float* __restrict__ lng, const float* __restrict__ lnb,
                         const float* __restrict__ fc2, float* __restrict__ gate)
{
    __shared__ float red[256];
    __shared__ float hsh[256];
    int b = blockIdx.x, o = threadIdx.x;
    const float* g = gf + b * 256;
    float acc = 0.f;
    for (int c = 0; c < 256; ++c) acc += g[c] * fc1[o * 256 + c];
    red[o] = acc; __syncthreads();
    for (int st = 128; st > 0; st >>= 1) { if (o < st) red[o] += red[o + st]; __syncthreads(); }
    float mean = red[0] * (1.f / 256.f); __syncthreads();
    float dv = acc - mean;
    red[o] = dv * dv; __syncthreads();
    for (int st = 128; st > 0; st >>= 1) { if (o < st) red[o] += red[o + st]; __syncthreads(); }
    float var = red[0] * (1.f / 256.f);
    float h = dv * rsqrtf(var + 1e-5f) * lng[o] + lnb[o];
    hsh[o] = h > 0.f ? h : 0.f;
    __syncthreads();
    float ga = 0.f;
    for (int c = 0; c < 256; ++c) ga += hsh[c] * fc2[o * 256 + c];
    gate[b * 256 + o] = ga;
}

// ---------------- weight prep: bf16 conversion + gate folding ----------------
__global__ void prep_weights(const float* __restrict__ nlb1W, const float* __restrict__ nlb2W,
                             const float* __restrict__ mlpW, const float* __restrict__ gate,
                             unsigned short* __restrict__ nlbW, unsigned short* __restrict__ mlpWb)
{
    size_t idx = (size_t)blockIdx.x * 256 + threadIdx.x;        // 16*65536 total
    if (idx < (size_t)12 * 65536) {
        int k = (int)(idx >> 16); size_t r = idx & 65535;
        float w = (k < 6) ? nlb1W[(size_t)k * 65536 + r] : nlb2W[(size_t)(k - 6) * 65536 + r];
        nlbW[idx] = f2bf(w);
    } else {
        size_t i2 = idx - (size_t)12 * 65536;                   // [4][256][256]
        int b = (int)(i2 >> 16); size_t rc = i2 & 65535; int c = (int)(rc & 255);
        mlpWb[i2] = f2bf(mlpW[rc] * gate[b * 256 + c]);
    }
}

// ---------------- gated pointwise conv + BN + ReLU (WMMA) ----------------
__global__ void mlp_kernel(const float* __restrict__ feat, float* __restrict__ out,
                           const unsigned short* __restrict__ Wb, const float* __restrict__ bn)
{
    extern __shared__ unsigned short smem[];                    // xs_t[128][256] bf16
    unsigned short* xs_t = smem;
    int tid = threadIdx.x, lane = tid & 31, wave = tid >> 5;
    const int lm = lane & 15;
    int nb = blockIdx.x, b = blockIdx.y;
    size_t nbase = (size_t)nb * 128;
    const float* fb = feat + (size_t)b * 256 * 16384;
    for (int idx = tid; idx < 256 * 128; idx += 256) {
        int c = idx >> 7, j = idx & 127;
        xs_t[j * 256 + c] = f2bf(fb[(size_t)c * 16384 + nbase + j]);
    }
    __syncthreads();
    const unsigned short* W = Wb + (size_t)b * 65536;

#pragma unroll 1
    for (int ph = 0; ph < 2; ++ph) {
        v8f acc[2][4];
#pragma unroll
        for (int oi = 0; oi < 2; ++oi)
#pragma unroll
            for (int pi = 0; pi < 4; ++pi) acc[oi][pi] = (v8f){};

#pragma unroll 2
        for (int kk = 0; kk < 8; ++kk) {
            v16bf a0 = load_a_frag(W, (wave * 2 + 0) * 16 + lm, 256, kk * 32, lane);
            v16bf a1 = load_a_frag(W, (wave * 2 + 1) * 16 + lm, 256, kk * 32, lane);
#pragma unroll
            for (int pi = 0; pi < 4; ++pi) {
                v16bf bv = load_b_frag(xs_t, (ph * 4 + pi) * 16 + lm, 256, kk * 32, lane);
                acc[0][pi] = wmma_bf16(a0, bv, acc[0][pi]);
                acc[1][pi] = wmma_bf16(a1, bv, acc[1][pi]);
            }
        }

#pragma unroll
        for (int oi = 0; oi < 2; ++oi) {
            int ot = wave * 2 + oi;
            int obase = ot * 16 + (lane >> 4) * 8;
            float sc[8], sh[8];
#pragma unroll
            for (int r = 0; r < 8; ++r) {
                int o = obase + r;
                float g = bn[o], bb = bn[256 + o], mm = bn[512 + o], vv = bn[768 + o];
                float is = rsqrtf(vv + 1e-5f);
                sc[r] = g * is; sh[r] = bb - g * mm * is;
            }
#pragma unroll
            for (int pi = 0; pi < 4; ++pi) {
                int n = (ph * 4 + pi) * 16 + lm;
#pragma unroll
                for (int r = 0; r < 8; ++r) {
                    float y = acc[oi][pi][r] * sc[r] + sh[r];
                    out[((size_t)(b * 256 + obase + r)) * 16384 + nbase + n] = y > 0.f ? y : 0.f;
                }
            }
        }
    }
}

// ---------------- fused axial attention stage (per (b,row) workgroup) ----------------
// in  : x[b][c][p][q] f32, attention over q (row length 128)
// out : transpose_out ? out[b][c][q][p] : out[b][c][p][q]   (f32)
__global__ void axial_kernel(const float* __restrict__ xin, float* __restrict__ out,
                             const unsigned short* __restrict__ Wq,
                             const unsigned short* __restrict__ Wk,
                             const unsigned short* __restrict__ Wv,
                             const float* __restrict__ bnq, const float* __restrict__ bnk,
                             const float* __restrict__ bnv, int transpose_out)
{
    extern __shared__ unsigned short smem[];      // 192 KB dynamic LDS
    unsigned short* xs_t  = smem;                 // [128][256] bf16 input row (pos-major)
    unsigned short* qt    = smem + 128 * 256;     // [128][256] q (pos-major) -> later vs[256][128]
    unsigned short* kt    = smem + 2 * 128 * 256; // [128][256] k (pos-major) -> later probs[128][128]
    unsigned short* vs    = qt;
    unsigned short* probs = kt;

    int tid = threadIdx.x, lane = tid & 31, wave = tid >> 5;
    const int lm = lane & 15;
    int p = blockIdx.x, b = blockIdx.y;

    // phase 1: stage the row into LDS (bf16). Row is read fully before any global write,
    // so even stages may run in-place on the same buffer.
    const float* xrow = xin + ((size_t)(b * 256)) * 128 * 128 + (size_t)p * 128;
    for (int idx = tid; idx < 256 * 128; idx += 256) {
        int c = idx >> 7, j = idx & 127;
        xs_t[j * 256 + c] = f2bf(xrow[(size_t)c * 16384 + j]);
    }
    __syncthreads();

    // phase 2: Q, K projections (conv + BN + ReLU), stored [pos][ch]
    project(Wq, bnq, xs_t, qt, true, wave, lane);
    project(Wk, bnk, xs_t, kt, true, wave, lane);
    __syncthreads();

    // phase 3: scores[i][j] = sum_c q[c,i]*k[c,j]; wave w owns rows i in [16w,16w+16).
    // Fully unrolled halves so sacc[] is only ever indexed by constants.
    v8f sacc[8];
    int ibase = wave * 16;
#pragma unroll
    for (int jh = 0; jh < 2; ++jh) {
        v8f acc[4];
#pragma unroll
        for (int ji = 0; ji < 4; ++ji) acc[ji] = (v8f){};
#pragma unroll 2
        for (int kk = 0; kk < 8; ++kk) {
            v16bf a = load_a_frag(qt, ibase + lm, 256, kk * 32, lane);
#pragma unroll
            for (int ji = 0; ji < 4; ++ji) {
                v16bf bv = load_b_frag(kt, (jh * 4 + ji) * 16 + lm, 256, kk * 32, lane);
                acc[ji] = wmma_bf16(a, bv, acc[ji]);
            }
        }
#pragma unroll
        for (int ji = 0; ji < 4; ++ji) sacc[jh * 4 + ji] = acc[ji];
    }
    __syncthreads();   // all waves done reading qt/kt; regions may be recycled

    // in-register softmax: lane holds rows m=(lane>>4)*8+r, col j=jt*16+(lane&15).
    // Row reduction: across 8 jt locally + xor-shuffles within the 16-lane group.
    float mx[8], siv[8];
#pragma unroll
    for (int r = 0; r < 8; ++r) {
        float m = -3.4e38f;
#pragma unroll
        for (int jt = 0; jt < 8; ++jt) m = fmaxf(m, sacc[jt][r]);
        for (int msk = 1; msk < 16; msk <<= 1) m = fmaxf(m, __shfl_xor(m, msk, 32));
        mx[r] = m;
    }
#pragma unroll
    for (int r = 0; r < 8; ++r) {
        float s = 0.f;
#pragma unroll
        for (int jt = 0; jt < 8; ++jt) {
            float e = __expf(sacc[jt][r] - mx[r]);
            sacc[jt][r] = e; s += e;
        }
        for (int msk = 1; msk < 16; msk <<= 1) s += __shfl_xor(s, msk, 32);
        siv[r] = 1.f / s;
    }
#pragma unroll
    for (int jt = 0; jt < 8; ++jt) {
        int j = jt * 16 + lm;
#pragma unroll
        for (int r = 0; r < 8; ++r) {
            int i = ibase + (lane >> 4) * 8 + r;
            probs[i * 128 + j] = f2bf(sacc[jt][r] * siv[r]);
        }
    }

    // V projection into vs[ch][pos] (recycled q region); xs_t still live
    project(Wv, bnv, xs_t, vs, false, wave, lane);
    __syncthreads();

    // phase 4: out[c][i] = sum_j vs[c][j] * probs[i][j]; 2 c-tiles x 4 i-tiles per pass
#pragma unroll 1
    for (int ih = 0; ih < 2; ++ih) {
        v8f acc[2][4];
#pragma unroll
        for (int ci = 0; ci < 2; ++ci)
#pragma unroll
            for (int ii = 0; ii < 4; ++ii) acc[ci][ii] = (v8f){};

#pragma unroll 2
        for (int kk = 0; kk < 4; ++kk) {
            v16bf a0 = load_a_frag(vs, (wave * 2 + 0) * 16 + lm, 128, kk * 32, lane);
            v16bf a1 = load_a_frag(vs, (wave * 2 + 1) * 16 + lm, 128, kk * 32, lane);
#pragma unroll
            for (int ii = 0; ii < 4; ++ii) {
                v16bf bv = load_b_frag(probs, (ih * 4 + ii) * 16 + lm, 128, kk * 32, lane);
                acc[0][ii] = wmma_bf16(a0, bv, acc[0][ii]);
                acc[1][ii] = wmma_bf16(a1, bv, acc[1][ii]);
            }
        }

#pragma unroll
        for (int ci = 0; ci < 2; ++ci) {
            int cbase = (wave * 2 + ci) * 16 + (lane >> 4) * 8;
#pragma unroll
            for (int ii = 0; ii < 4; ++ii) {
                int i = (ih * 4 + ii) * 16 + lm;
#pragma unroll
                for (int r = 0; r < 8; ++r) {
                    int c = cbase + r;
                    size_t addr = transpose_out
                        ? (((size_t)(b * 256 + c)) * 128 + i) * 128 + p
                        : (((size_t)(b * 256 + c)) * 128 + p) * 128 + i;
                    out[addr] = acc[ci][ii][r];
                }
            }
        }
    }
}

// ---------------- host ----------------
extern "C" void kernel_launch(void* const* d_in, const int* in_sizes, int n_in,
                              void* d_out, int out_size, void* d_ws, size_t ws_size,
                              hipStream_t stream)
{
    (void)in_sizes; (void)n_in; (void)out_size; (void)ws_size;
    const float* feature    = (const float*)d_in[0];
    const float* gcb_mlp1_W = (const float*)d_in[1];
    const float* gcb_mlp1_bn= (const float*)d_in[2];
    const float* gcb_fc1_W  = (const float*)d_in[3];
    const float* gcb_ln_g   = (const float*)d_in[4];
    const float* gcb_ln_b   = (const float*)d_in[5];
    const float* gcb_fc2_W  = (const float*)d_in[6];
    const float* mlp_W      = (const float*)d_in[7];
    const float* mlp_bn     = (const float*)d_in[8];
    const float* nlb1_W     = (const float*)d_in[9];
    const float* nlb1_bn    = (const float*)d_in[10];
    const float* nlb2_W     = (const float*)d_in[11];
    const float* nlb2_bn    = (const float*)d_in[12];
    float* out = (float*)d_out;

    char* ws = (char*)d_ws;
    float* buf0 = (float*)ws;                                       // 64 MB ping
    size_t off = (size_t)4 * 256 * 16384 * 4;
    float* att  = (float*)(ws + off); off += (size_t)4 * 16384 * 4;
    float* gf   = (float*)(ws + off); off += 4 * 256 * 4;
    float* gate = (float*)(ws + off); off += 4 * 256 * 4;
    unsigned short* nlbW  = (unsigned short*)(ws + off); off += (size_t)12 * 65536 * 2;
    unsigned short* mlpWb = (unsigned short*)(ws + off); off += (size_t)4 * 65536 * 2;

    const int ldsA = 3 * 128 * 256 * 2;   // 192 KB for axial
    const int ldsM = 128 * 256 * 2;       // 64 KB for mlp
    (void)hipFuncSetAttribute((const void*)axial_kernel,
                              hipFuncAttributeMaxDynamicSharedMemorySize, ldsA);
    (void)hipFuncSetAttribute((const void*)mlp_kernel,
                              hipFuncAttributeMaxDynamicSharedMemorySize, ldsM);

    gcb_logits<<<dim3(64, 4), 256, 0, stream>>>(feature, gcb_mlp1_W, gcb_mlp1_bn, att);
    gcb_softmax<<<4, 256, 0, stream>>>(att);
    gcb_gf<<<dim3(256, 4), 256, 0, stream>>>(feature, att, gf);
    gcb_gate<<<4, 256, 0, stream>>>(gf, gcb_fc1_W, gcb_ln_g, gcb_ln_b, gcb_fc2_W, gate);
    prep_weights<<<4096, 256, 0, stream>>>(nlb1_W, nlb2_W, mlp_W, gate, nlbW, mlpWb);

    mlp_kernel<<<dim3(128, 4), 256, ldsM, stream>>>(feature, buf0, mlpWb, mlp_bn);

    // NLB1 stage1 (attend j, write transposed), stage2 (attend i, in-place straight)
    axial_kernel<<<dim3(128, 4), 256, ldsA, stream>>>(buf0, out,
        nlbW + 0 * 65536, nlbW + 1 * 65536, nlbW + 2 * 65536,
        nlb1_bn + 0 * 1024, nlb1_bn + 1 * 1024, nlb1_bn + 2 * 1024, 1);
    axial_kernel<<<dim3(128, 4), 256, ldsA, stream>>>(out, out,
        nlbW + 3 * 65536, nlbW + 4 * 65536, nlbW + 5 * 65536,
        nlb1_bn + 3 * 1024, nlb1_bn + 4 * 1024, nlb1_bn + 5 * 1024, 0);
    // NLB2 stage1 / stage2
    axial_kernel<<<dim3(128, 4), 256, ldsA, stream>>>(out, buf0,
        nlbW + 6 * 65536, nlbW + 7 * 65536, nlbW + 8 * 65536,
        nlb2_bn + 0 * 1024, nlb2_bn + 1 * 1024, nlb2_bn + 2 * 1024, 1);
    axial_kernel<<<dim3(128, 4), 256, ldsA, stream>>>(buf0, out,
        nlbW + 9 * 65536, nlbW + 10 * 65536, nlbW + 11 * 65536,
        nlb2_bn + 3 * 1024, nlb2_bn + 4 * 1024, nlb2_bn + 5 * 1024, 0);
}
